// network_82635170775350
// MI455X (gfx1250) — matrix operand order repaired and checked by reading, not compiled, not run on hardware
//
#include <hip/hip_runtime.h>

#define Bn   16384
#define NAg  5
#define Dd   128
#define LS1d 128
#define Hh   128
#define Ao   64
#define G4   512   // 4*H
#define K6   768   // 6*H
#define SPITCH 136 // padded LDS row pitch in halves (128 + 8) -> conflict-free

#define DEVINLINE __device__ __attribute__((always_inline)) inline

typedef __attribute__((ext_vector_type(16))) _Float16 v16h;
typedef __attribute__((ext_vector_type(8)))  _Float16 v8h;
typedef __attribute__((ext_vector_type(8)))  float    v8f;
typedef __attribute__((ext_vector_type(4)))  unsigned int u32x4;
typedef __attribute__((ext_vector_type(4)))  int i32x4;
typedef __attribute__((ext_vector_type(8)))  int i32x8;

#if defined(__AMDGCN__) && __has_builtin(__builtin_amdgcn_tensor_load_to_lds)
#define HAVE_TDM 1
#else
#define HAVE_TDM 0
#endif

__constant__ int dPERM[5][5] = {
    {1,2,3,4,0},{0,2,3,4,1},{0,1,3,4,2},{0,1,2,4,3},{0,1,2,3,4}};
__constant__ int dTIDX[5][5] = {
    {4,0,0,0,0},{0,4,1,1,1},{1,1,4,2,2},{2,2,2,4,3},{3,3,3,3,4}};

// ---------------- WMMA fragment helpers (gfx1250 wave32) ----------------
// A (16x32 f16, M = lane&15): VGPR0..3 hold K = 8*hi .. 8*hi+7 (2/VGPR),
// VGPR4..7 hold K = 16+8*hi .. 16+8*hi+7, hi = lane>>4.
DEVINLINE v16h load_a_f16(const _Float16* base, int stride, int kbase, int lane) {
    const _Float16* p = base + (size_t)(lane & 15) * stride + kbase + ((lane >> 4) << 3);
    v8h lo = *(const v8h*)(p);
    v8h hi = *(const v8h*)(p + 16);
    v16h r;
#pragma unroll
    for (int i = 0; i < 8; ++i) { r[i] = lo[i]; r[i + 8] = hi[i]; }
    return r;
}

// Same fragment, converting from an f32 source on the fly.
DEVINLINE v16h load_a_f32cvt(const float* base, int stride, int kbase, int lane) {
    const float* p = base + (size_t)(lane & 15) * stride + kbase + ((lane >> 4) << 3);
    v16h r;
#pragma unroll
    for (int i = 0; i < 8; ++i) { r[i] = (_Float16)p[i]; r[i + 8] = (_Float16)p[16 + i]; }
    return r;
}

// B (32x16 f16): lane n = lane&15 holds row n of W (B = W^T); same K layout as A:
// K = 8*hi .. 8*hi+7 then 16+8*hi .. 16+8*hi+7, hi = lane>>4.
DEVINLINE v16h load_b_f16(const _Float16* base, int stride, int kbase, int lane) {
    const _Float16* p = base + (size_t)(lane & 15) * stride + kbase + ((lane >> 4) << 3);
    v8h lo = *(const v8h*)(p);
    v8h hi = *(const v8h*)(p + 16);
    v16h r;
#pragma unroll
    for (int i = 0; i < 8; ++i) { r[i] = lo[i]; r[i + 8] = hi[i]; }
    return r;
}

DEVINLINE v8f wmma16(v16h a, v16h b, v8f c) {
    return __builtin_amdgcn_wmma_f32_16x16x32_f16(false, a, false, b, (short)0, c, false, false);
}

DEVINLINE float sigm(float x) { return 1.0f / (1.0f + __expf(-x)); }

// ---------------- TDM staging: 32x128 f16 tile -> padded LDS ----------------
#if HAVE_TDM
// D# per CDNA5 ISA §8: 2D tile, data_size=2B (code 1), tile 128x32,
// tensor_dim0_stride = 128 elems, pad_enable: 4 dwords every 64 dwords
// (pad_interval code 5, pad_amount code 3) -> LDS pitch 272B = SPITCH halves.
DEVINLINE void tdm_load_tile32x128(unsigned lds_off, const _Float16* gsrc) {
    unsigned long long ga = (unsigned long long)(size_t)gsrc;
    u32x4 g0;
    g0[0] = 1u;                                   // count=1, user descriptor
    g0[1] = lds_off;                              // LDS byte address
    g0[2] = (unsigned)ga;                         // global_addr lo
    g0[3] = (unsigned)(ga >> 32) | (2u << 30);    // global_addr hi | type=2
    i32x8 g1;
    g1[0] = (1 << 16) | (1 << 20) | (5 << 22) | (3 << 25); // 2B, pad 4dw/64dw
    g1[1] = (int)((128u & 0xffffu) << 16);        // tensor_dim0[15:0] in [31:16]
    g1[2] = (int)((32u & 0xffffu) << 16);         // dim0 hi=0 | tensor_dim1 lo
    g1[3] = (int)((128u & 0xffffu) << 16);        // dim1 hi=0 | tile_dim0=128
    g1[4] = 32;                                   // tile_dim1=32, tile_dim2=0
    g1[5] = 128;                                  // tensor_dim0_stride lo32
    g1[6] = 0;
    g1[7] = 0;
    i32x4 z4 = {0, 0, 0, 0};
#if defined(__clang_major__) && (__clang_major__ >= 23)
    i32x8 z8 = {0, 0, 0, 0, 0, 0, 0, 0};
    __builtin_amdgcn_tensor_load_to_lds(g0, g1, z4, z4, z8, 0);
#else
    __builtin_amdgcn_tensor_load_to_lds(g0, g1, z4, z4, 0);
#endif
}
#endif

// Fallback / host-pass staging: cooperative copy, 256 threads x 16B.
DEVINLINE void coop_stage32x128(_Float16* sdst, const _Float16* gsrc, int tid) {
    int row = tid >> 3, c = tid & 7;
    *(v8h*)(sdst + row * SPITCH + c * 8) = *(const v8h*)(gsrc + (size_t)row * 128 + c * 8);
}

// ---------------- small prep kernels ----------------
__global__ void cvt_kernel(const float* __restrict__ s, _Float16* __restrict__ d, size_t n) {
    size_t i = (size_t)blockIdx.x * 256 + threadIdx.x;
    size_t stride = (size_t)gridDim.x * 256;
    for (; i < n; i += stride) d[i] = (_Float16)s[i];
}

__global__ void bias_sum_kernel(const float* __restrict__ a, const float* __restrict__ b,
                                float* __restrict__ d, size_t n) {
    size_t i = (size_t)blockIdx.x * 256 + threadIdx.x;
    size_t stride = (size_t)gridDim.x * 256;
    for (; i < n; i += stride) d[i] = a[i] + b[i];
}

// ---------------- obs = input · d1_wᵀ + d1_b (stored f16) ----------------
__global__ __launch_bounds__(256) void obs_kernel(const float* __restrict__ input,
                                                  const _Float16* __restrict__ d1w,
                                                  const float* __restrict__ d1b,
                                                  _Float16* __restrict__ obs_h) {
    int lane = threadIdx.x & 31, wave = threadIdx.x >> 5;
    int m0 = blockIdx.x * 16;
    int n  = blockIdx.y;
    int jcol = wave * 16;
    const float* abase = input + ((size_t)m0 * NAg + n) * Dd;   // row stride NAg*Dd
    const _Float16* bbase = d1w + (size_t)jcol * Dd;
    v8f acc = {};
#pragma unroll
    for (int k = 0; k < 4; ++k) {
        v16h a = load_a_f32cvt(abase, NAg * Dd, k * 32, lane);
        v16h b = load_b_f16(bbase, Dd, k * 32, lane);
        acc = wmma16(a, b, acc);
    }
    int col = jcol + (lane & 15);
    float bias = d1b[col];
    int rbase = (lane >> 4) << 3;
#pragma unroll
    for (int r = 0; r < 8; ++r) {
        obs_h[((size_t)n * Bn + m0 + rbase + r) * LS1d + col] = (_Float16)(acc[r] + bias);
    }
}

// ---------------- LSTM tile core: 32 rows/block, 2 M-tiles per wave ----------
// A-fragments come from padded LDS tiles; each weight (B) fragment is reused
// across both M-tiles. One wave owns 16 gate columns of all four gates, so the
// whole cell update stays in registers. always_inline: specialize per kernel,
// fold has_h / null-output branches, schedule loads across the whole body.
DEVINLINE void lstm_tile32(const _Float16* sX, const _Float16* sH, int has_h,
                           const _Float16* Wih, const _Float16* Whh,
                           const float* bsum, const float* cprev,
                           float* cout, float* hout_f32, _Float16* hout_f16,
                           int m0blk, int lane, int wave) {
    int jcol = wave * 16;
    v8f zero = {};
    v8f acc[2][4];
#pragma unroll
    for (int mt = 0; mt < 2; ++mt)
#pragma unroll
        for (int g = 0; g < 4; ++g) acc[mt][g] = zero;

#pragma unroll
    for (int k = 0; k < 4; ++k) {
        v16h a0 = load_a_f16(sX, SPITCH, k * 32, lane);
        v16h a1 = load_a_f16(sX + 16 * SPITCH, SPITCH, k * 32, lane);
#pragma unroll
        for (int g = 0; g < 4; ++g) {
            v16h b = load_b_f16(Wih + (size_t)(g * Hh + jcol) * LS1d, LS1d, k * 32, lane);
            acc[0][g] = wmma16(a0, b, acc[0][g]);
            acc[1][g] = wmma16(a1, b, acc[1][g]);
        }
    }
    if (has_h) {                                  // uniform branch; EXEC all-1s
#pragma unroll
        for (int k = 0; k < 4; ++k) {
            v16h a0 = load_a_f16(sH, SPITCH, k * 32, lane);
            v16h a1 = load_a_f16(sH + 16 * SPITCH, SPITCH, k * 32, lane);
#pragma unroll
            for (int g = 0; g < 4; ++g) {
                v16h b = load_b_f16(Whh + (size_t)(g * Hh + jcol) * Hh, Hh, k * 32, lane);
                acc[0][g] = wmma16(a0, b, acc[0][g]);
                acc[1][g] = wmma16(a1, b, acc[1][g]);
            }
        }
    }
    int col = jcol + (lane & 15);
    float bi = bsum[col], bf = bsum[128 + col], bg = bsum[256 + col], bo = bsum[384 + col];
    int rbase = (lane >> 4) << 3;
#pragma unroll
    for (int mt = 0; mt < 2; ++mt) {
#pragma unroll
        for (int r = 0; r < 8; ++r) {
            size_t idx = (size_t)(m0blk + mt * 16 + rbase + r) * Hh + col;
            float cp = cprev ? cprev[idx] : 0.0f;
            float iv = sigm(acc[mt][0][r] + bi);
            float fv = sigm(acc[mt][1][r] + bf);
            float gv = tanhf(acc[mt][2][r] + bg);
            float ov = sigm(acc[mt][3][r] + bo);
            float c2 = fv * cp + iv * gv;
            float h2 = ov * tanhf(c2);
            cout[idx] = c2;
            if (hout_f32) hout_f32[idx] = h2;
            if (hout_f16) hout_f16[idx] = (_Float16)h2;
        }
    }
}

// ---------------- memory-LSTM (writes h_n / c_n outputs) ----------------
__global__ __launch_bounds__(256) void mem_kernel(const _Float16* __restrict__ obs_h,
                                                  const _Float16* __restrict__ h0_h,
                                                  const float* __restrict__ c0,
                                                  const _Float16* __restrict__ Wih,
                                                  const _Float16* __restrict__ Whh,
                                                  const float* __restrict__ bsum,
                                                  float* __restrict__ hn_out,
                                                  float* __restrict__ cn_out,
                                                  _Float16* __restrict__ memh_h) {
    __shared__ __align__(16) _Float16 sX[32 * SPITCH];
    __shared__ __align__(16) _Float16 sH[32 * SPITCH];
    int n = blockIdx.y;
    int m0 = blockIdx.x * 32;
    int lane = threadIdx.x & 31, wave = threadIdx.x >> 5;
    const _Float16* xg = obs_h + ((size_t)n * Bn + m0) * LS1d;
    const _Float16* hg = h0_h + ((size_t)n * Bn + m0) * Hh;
#if HAVE_TDM
    if (threadIdx.x < 32) {
        tdm_load_tile32x128((unsigned)(unsigned long long)(void*)sX, xg);
        tdm_load_tile32x128((unsigned)(unsigned long long)(void*)sH, hg);
        __builtin_amdgcn_s_wait_tensorcnt(0);
    }
#else
    coop_stage32x128(sX, xg, threadIdx.x);
    coop_stage32x128(sH, hg, threadIdx.x);
#endif
    __syncthreads();
    lstm_tile32(sX, sH, 1,
                Wih + (size_t)n * G4 * LS1d, Whh + (size_t)n * G4 * Hh,
                bsum + (size_t)n * G4,
                c0 + (size_t)n * Bn * Hh,
                cn_out + (size_t)n * Bn * Hh,
                hn_out + (size_t)n * Bn * Hh,
                memh_h + (size_t)n * Bn * Hh,
                m0, lane, wave);
}

// ---------------- communication-LSTM step t (sequential scan) ----------------
__global__ __launch_bounds__(256) void com_kernel(const _Float16* __restrict__ obs_h,
                                                  const _Float16* ys_h,   // read t-1 / write t
                                                  float* comc,
                                                  const _Float16* __restrict__ Wih,
                                                  const _Float16* __restrict__ Whh,
                                                  const float* __restrict__ bsum,
                                                  int t) {
    __shared__ __align__(16) _Float16 sX[32 * SPITCH];
    __shared__ __align__(16) _Float16 sH[32 * SPITCH];
    int i = blockIdx.y;
    int m0 = blockIdx.x * 32;
    int lane = threadIdx.x & 31, wave = threadIdx.x >> 5;
    int src = dPERM[i][t];
    const _Float16* xg = obs_h + ((size_t)src * Bn + m0) * LS1d;
    const _Float16* hg = (t == 0) ? (const _Float16*)0
                                  : ys_h + (((size_t)(t - 1) * NAg + i) * Bn + m0) * Hh;
#if HAVE_TDM
    if (threadIdx.x < 32) {
        tdm_load_tile32x128((unsigned)(unsigned long long)(void*)sX, xg);
        if (t > 0) tdm_load_tile32x128((unsigned)(unsigned long long)(void*)sH, hg);
        __builtin_amdgcn_s_wait_tensorcnt(0);
    }
#else
    coop_stage32x128(sX, xg, threadIdx.x);
    if (t > 0) coop_stage32x128(sH, hg, threadIdx.x);
#endif
    __syncthreads();
    const float* cp = (t == 0) ? (const float*)0 : comc + (size_t)i * Bn * Hh;
    lstm_tile32(sX, sH, (t > 0),
                Wih + (size_t)i * G4 * LS1d, Whh + (size_t)i * G4 * Hh,
                bsum + (size_t)i * G4,
                cp,
                comc + (size_t)i * Bn * Hh,
                (float*)0,
                (_Float16*)ys_h + (((size_t)t * NAg + i) * Bn) * Hh,
                m0, lane, wave);
}

// ---------------- outs = cat(sel, mem_h) · d2_wᵀ + d2_b ----------------
__global__ __launch_bounds__(256) void out_kernel(const _Float16* __restrict__ ys_h,
                                                  const _Float16* __restrict__ memh_h,
                                                  const _Float16* __restrict__ d2w,
                                                  const float* __restrict__ d2b,
                                                  float* __restrict__ out) {
    int i = blockIdx.y;
    int lane = threadIdx.x & 31, wave = threadIdx.x >> 5;
    int mt = wave >> 2, nt = wave & 3;
    int m0 = blockIdx.x * 32 + mt * 16;
    int jcol = nt * 16;
    v8f acc = {};
#pragma unroll
    for (int s = 0; s < 6; ++s) {
        const _Float16* abase;
        if (s < 5) abase = ys_h + (((size_t)dTIDX[i][s] * NAg + s) * Bn + m0) * Hh;
        else       abase = memh_h + ((size_t)i * Bn + m0) * Hh;
#pragma unroll
        for (int k = 0; k < 4; ++k) {
            v16h a = load_a_f16(abase, Hh, k * 32, lane);
            v16h b = load_b_f16(d2w + (size_t)jcol * K6, K6, s * 128 + k * 32, lane);
            acc = wmma16(a, b, acc);
        }
    }
    int col = jcol + (lane & 15);
    float bias = d2b[col];
    int rbase = (lane >> 4) << 3;
#pragma unroll
    for (int r = 0; r < 8; ++r) {
        out[((size_t)i * Bn + m0 + rbase + r) * Ao + col] = acc[r] + bias;
    }
}

// ---------------- host side ----------------
extern "C" void kernel_launch(void* const* d_in, const int* in_sizes, int n_in,
                              void* d_out, int out_size, void* d_ws, size_t ws_size,
                              hipStream_t stream) {
    (void)in_sizes; (void)n_in; (void)out_size; (void)ws_size;
    const float* input = (const float*)d_in[0];
    const float* h0    = (const float*)d_in[1];
    const float* c0    = (const float*)d_in[2];
    const float* d1_w  = (const float*)d_in[3];
    const float* d1_b  = (const float*)d_in[4];
    const float* d2_w  = (const float*)d_in[5];
    const float* d2_b  = (const float*)d_in[6];
    const float* mWih  = (const float*)d_in[7];
    const float* mWhh  = (const float*)d_in[8];
    const float* mbih  = (const float*)d_in[9];
    const float* mbhh  = (const float*)d_in[10];
    const float* cWih  = (const float*)d_in[11];
    const float* cWhh  = (const float*)d_in[12];
    const float* cbih  = (const float*)d_in[13];
    const float* cbhh  = (const float*)d_in[14];

    float* out    = (float*)d_out;
    float* hn_out = out + (size_t)NAg * Bn * Ao;
    float* cn_out = hn_out + (size_t)NAg * Bn * Hh;

    // workspace carve-out (256B aligned segments), total ~213 MB
    char* p = (char*)d_ws;
    auto take = [&](size_t bytes) -> void* {
        void* r = (void*)p;
        p += (bytes + 255) & ~(size_t)255;
        return r;
    };
    _Float16* d1w_h  = (_Float16*)take((size_t)LS1d * Dd * 2);
    _Float16* d2w_h  = (_Float16*)take((size_t)Ao * K6 * 2);
    _Float16* mWih_h = (_Float16*)take((size_t)NAg * G4 * LS1d * 2);
    _Float16* mWhh_h = (_Float16*)take((size_t)NAg * G4 * Hh * 2);
    _Float16* cWih_h = (_Float16*)take((size_t)NAg * G4 * LS1d * 2);
    _Float16* cWhh_h = (_Float16*)take((size_t)NAg * G4 * Hh * 2);
    float*    mb     = (float*)take((size_t)NAg * G4 * 4);
    float*    cb     = (float*)take((size_t)NAg * G4 * 4);
    _Float16* obs_h  = (_Float16*)take((size_t)NAg * Bn * LS1d * 2);
    _Float16* h0_h   = (_Float16*)take((size_t)NAg * Bn * Hh * 2);
    _Float16* memh_h = (_Float16*)take((size_t)NAg * Bn * Hh * 2);
    _Float16* ys_h   = (_Float16*)take((size_t)5 * NAg * Bn * Hh * 2);
    float*    comc   = (float*)take((size_t)NAg * Bn * Hh * 4);

    auto cvt = [&](const float* s, _Float16* d, size_t n) {
        int blocks = (int)((n + 255) / 256);
        if (blocks > 4096) blocks = 4096;
        cvt_kernel<<<blocks, 256, 0, stream>>>(s, d, n);
    };
    cvt(d1_w, d1w_h, (size_t)LS1d * Dd);
    cvt(d2_w, d2w_h, (size_t)Ao * K6);
    cvt(mWih, mWih_h, (size_t)NAg * G4 * LS1d);
    cvt(mWhh, mWhh_h, (size_t)NAg * G4 * Hh);
    cvt(cWih, cWih_h, (size_t)NAg * G4 * LS1d);
    cvt(cWhh, cWhh_h, (size_t)NAg * G4 * Hh);
    cvt(h0, h0_h, (size_t)NAg * Bn * Hh);
    bias_sum_kernel<<<10, 256, 0, stream>>>(mbih, mbhh, mb, (size_t)NAg * G4);
    bias_sum_kernel<<<10, 256, 0, stream>>>(cbih, cbhh, cb, (size_t)NAg * G4);

    dim3 blk(256);
    obs_kernel<<<dim3(Bn / 16, NAg), blk, 0, stream>>>(input, d1w_h, d1_b, obs_h);
    mem_kernel<<<dim3(Bn / 32, NAg), blk, 0, stream>>>(obs_h, h0_h, c0, mWih_h, mWhh_h, mb,
                                                       hn_out, cn_out, memh_h);
    for (int t = 0; t < 5; ++t) {
        com_kernel<<<dim3(Bn / 32, NAg), blk, 0, stream>>>(obs_h, ys_h, comc,
                                                           cWih_h, cWhh_h, cb, t);
    }
    out_kernel<<<dim3(Bn / 32, NAg), blk, 0, stream>>>(ys_h, memh_h, d2w_h, d2_b, out);
}
